// FixedSafetyTokenSelector_32220844654684
// MI455X (gfx1250) — matrix operand order repaired and checked by reading, not compiled, not run on hardware
//
#include <hip/hip_runtime.h>
#include <hip/hip_bf16.h>
#include <math.h>

typedef __attribute__((ext_vector_type(16))) __bf16 v16bf;
typedef __attribute__((ext_vector_type(8)))  float  v8f;
typedef __attribute__((ext_vector_type(4)))  float  f4;

#define D_IN   768
#define H_DIM  384
#define S_LEN  4096
#define B_SZ   16
#define KSEL   40
#define NKSTEP (D_IN / 32)    // 24 k-steps of 32
#define NTILES (H_DIM / 16)   // 24 column tiles

// ---------------------------------------------------------------------------
// Kernel 1: repack W1 [768 x 384] f32 row-major -> bf16 WMMA B-fragments,
// layout [ks][t][lane][e] so one wave's 6 fragments per k-step are 6
// consecutive 1KB blocks (single base address + immediate offsets).
// 16-bit B layout: lanes 0-15 hold K={0..7,16..23}, lanes 16-31 hold K={8..15,24..31},
// column N = t*16 + (lane & 15).
// ---------------------------------------------------------------------------
__global__ void pack_w1(const float* __restrict__ W1, __bf16* __restrict__ Wp) {
    int id = blockIdx.x * blockDim.x + threadIdx.x;   // 0 .. 294911
    int e  = id & 15;
    int l  = (id >> 4) & 31;
    int r  = id >> 9;          // (ks*24 + t)
    int t  = r % NTILES;
    int ks = r / NTILES;
    int n  = t * 16 + (l & 15);
    int k  = ks * 32 + ((l & 16) >> 1) + (e & 7) + ((e & 8) << 1);
    Wp[id] = (__bf16)W1[k * H_DIM + n];
}

// ---------------------------------------------------------------------------
// Kernel 2: 32-row x 384-col tile per block. 8 waves = 2(M) x 4(N).
// Double-buffered bf16 WMMA GEMM (fp32 accum) -> LDS -> LayerNorm ->
// GELU(exact) -> W2 dot -> sigmoid -> scores.
// ---------------------------------------------------------------------------
__global__ __launch_bounds__(256) void gemm_ln_gelu_score(
    const float* __restrict__ F, const __bf16* __restrict__ Wp,
    const float* __restrict__ b1, const float* __restrict__ gamma,
    const float* __restrict__ beta, const float* __restrict__ W2,
    const float* __restrict__ b2, float* __restrict__ scores)
{
    __shared__ float hs[32][H_DIM + 4];
    __shared__ float red[32][8];
    __shared__ float red2[32][8];
    __shared__ float s_mean[32];
    __shared__ float s_rstd[32];

    const int tid  = threadIdx.x;
    const int lane = tid & 31;
    const int w    = tid >> 5;     // wave 0..7
    const int mt   = w >> 2;       // M-tile 0..1
    const int ng   = w & 3;        // N-group 0..3 (6 ntiles each)
    const int m0   = blockIdx.x * 32;

    // A-fragment lane mapping (16-bit A 16x32 layout):
    // row M = lane&15; lanes<16: K in {0..7}U{16..23}; lanes>=16: +8.
    const int arow  = m0 + mt * 16 + (lane & 15);
    const int kbase = (lane & 16) >> 1;            // 0 or 8
    const float* __restrict__ Arow = F + (size_t)arow * D_IN + kbase;

    // B fragment base for this wave: fragment (ks, j) at
    // Bp + ks*12288 + j*512 elements (j strides are 1KB immediates).
    const __bf16* __restrict__ Bp =
        Wp + (size_t)(ng * 6) * 512 + (size_t)lane * 16;

    v8f   acc[6] = {};
    v16bf aa[2];
    v16bf bb[2][6];

    // ---- prologue: load k-step 0 ----
    {
        f4 a0 = *(const f4*)(Arow + 0);
        f4 a1 = *(const f4*)(Arow + 4);
        f4 a2 = *(const f4*)(Arow + 16);
        f4 a3 = *(const f4*)(Arow + 20);
        v16bf af;
        af[0]  = (__bf16)a0[0]; af[1]  = (__bf16)a0[1];
        af[2]  = (__bf16)a0[2]; af[3]  = (__bf16)a0[3];
        af[4]  = (__bf16)a1[0]; af[5]  = (__bf16)a1[1];
        af[6]  = (__bf16)a1[2]; af[7]  = (__bf16)a1[3];
        af[8]  = (__bf16)a2[0]; af[9]  = (__bf16)a2[1];
        af[10] = (__bf16)a2[2]; af[11] = (__bf16)a2[3];
        af[12] = (__bf16)a3[0]; af[13] = (__bf16)a3[1];
        af[14] = (__bf16)a3[2]; af[15] = (__bf16)a3[3];
        aa[0] = af;
#pragma unroll
        for (int j = 0; j < 6; ++j)
            bb[0][j] = *(const v16bf*)(Bp + j * 512);
    }

    for (int ks = 0; ks < NKSTEP; ++ks) {
        const int cur = ks & 1;
        const int nxt = cur ^ 1;
        if (ks + 1 < NKSTEP) {
            // ---- prefetch next k-step into the other buffer ----
            const float* __restrict__ Ap = Arow + (ks + 1) * 32;
            f4 a0 = *(const f4*)(Ap + 0);
            f4 a1 = *(const f4*)(Ap + 4);
            f4 a2 = *(const f4*)(Ap + 16);
            f4 a3 = *(const f4*)(Ap + 20);
            const __bf16* __restrict__ Bk = Bp + (size_t)(ks + 1) * 12288;
#pragma unroll
            for (int j = 0; j < 6; ++j)
                bb[nxt][j] = *(const v16bf*)(Bk + j * 512);
            v16bf af;
            af[0]  = (__bf16)a0[0]; af[1]  = (__bf16)a0[1];
            af[2]  = (__bf16)a0[2]; af[3]  = (__bf16)a0[3];
            af[4]  = (__bf16)a1[0]; af[5]  = (__bf16)a1[1];
            af[6]  = (__bf16)a1[2]; af[7]  = (__bf16)a1[3];
            af[8]  = (__bf16)a2[0]; af[9]  = (__bf16)a2[1];
            af[10] = (__bf16)a2[2]; af[11] = (__bf16)a2[3];
            af[12] = (__bf16)a3[0]; af[13] = (__bf16)a3[1];
            af[14] = (__bf16)a3[2]; af[15] = (__bf16)a3[3];
            aa[nxt] = af;
        }
        // ---- 6 back-to-back WMMAs on current buffer (no A/B hazards) ----
#pragma unroll
        for (int j = 0; j < 6; ++j)
            acc[j] = __builtin_amdgcn_wmma_f32_16x16x32_bf16(
                false, aa[cur], false, bb[cur][j], (short)0, acc[j],
                false, false);
    }

    // C/D layout: lanes<16: N=lane, M=v; lanes>=16: N=lane-16, M=v+8.
    {
        const int rl = mt * 16 + ((lane & 16) >> 1); // +0 or +8
        const int cb = lane & 15;
#pragma unroll
        for (int j = 0; j < 6; ++j) {
            const int c = (ng * 6 + j) * 16 + cb;
#pragma unroll
            for (int v = 0; v < 8; ++v)
                hs[rl + v][c] = acc[j][v];
        }
    }
    __syncthreads();

    // LayerNorm + GELU + W2 dot: 8 threads per row, 48 cols each.
    const int row = tid >> 3;
    const int sub = tid & 7;
    {
        float sum = 0.f, sq = 0.f;
        for (int c = sub * 48; c < sub * 48 + 48; ++c) {
            float v = hs[row][c] + b1[c];
            hs[row][c] = v;
            sum += v; sq += v * v;
        }
        red[row][sub] = sum; red2[row][sub] = sq;
    }
    __syncthreads();
    if (sub == 0) {
        float s = 0.f, q = 0.f;
        for (int i = 0; i < 8; ++i) { s += red[row][i]; q += red2[row][i]; }
        float mean = s * (1.0f / H_DIM);
        float var  = q * (1.0f / H_DIM) - mean * mean;
        s_mean[row] = mean;
        s_rstd[row] = rsqrtf(var + 1e-5f);
    }
    __syncthreads();
    {
        const float mean = s_mean[row], rstd = s_rstd[row];
        float dot = 0.f;
        for (int c = sub * 48; c < sub * 48 + 48; ++c) {
            float v = (hs[row][c] - mean) * rstd * gamma[c] + beta[c];
            float g = 0.5f * v * (1.0f + erff(v * 0.70710678118654752f));
            dot += g * W2[c];
        }
        red[row][sub] = dot;
    }
    __syncthreads();
    if (sub == 0) {
        float s = 0.f;
        for (int i = 0; i < 8; ++i) s += red[row][i];
        s += b2[0];
        scores[m0 + row] = 1.0f / (1.0f + expf(-s));
    }
}

// ---------------------------------------------------------------------------
// Kernel 3: top-40 per batch (iterative argmax, jax tie-break: lower index).
// ---------------------------------------------------------------------------
__global__ __launch_bounds__(256) void topk40(
    const float* __restrict__ scores, int* __restrict__ idx_i,
    float* __restrict__ idx_f)
{
    __shared__ float sv[S_LEN];
    __shared__ float rv[256];
    __shared__ int   ri[256];
    const int b   = blockIdx.x;
    const int tid = threadIdx.x;
    const float* s = scores + (size_t)b * S_LEN;
    for (int i = tid; i < S_LEN; i += 256) sv[i] = s[i];
    __syncthreads();
    for (int k = 0; k < KSEL; ++k) {
        float best = -__builtin_inff();
        int   bi   = 0x7fffffff;
        for (int i = tid; i < S_LEN; i += 256) {
            float v = sv[i];
            if (v > best || (v == best && i < bi)) { best = v; bi = i; }
        }
        rv[tid] = best; ri[tid] = bi;
        __syncthreads();
        for (int off = 128; off > 0; off >>= 1) {
            if (tid < off) {
                float v = rv[tid + off]; int i2 = ri[tid + off];
                if (v > rv[tid] || (v == rv[tid] && i2 < ri[tid])) {
                    rv[tid] = v; ri[tid] = i2;
                }
            }
            __syncthreads();
        }
        if (tid == 0) {
            int win = ri[0];
            idx_i[b * KSEL + k] = win;
            idx_f[b * KSEL + k] = (float)win;
            sv[win] = -__builtin_inff();
        }
        __syncthreads();
    }
}

// ---------------------------------------------------------------------------
// Kernel 4: gather winning feature rows.
// ---------------------------------------------------------------------------
__global__ void gather_tokens(const float* __restrict__ F,
                              const int* __restrict__ idx,
                              float* __restrict__ out)
{
    const int bk = blockIdx.x;          // 0 .. B*KSEL-1
    const int b  = bk / KSEL;
    const int t  = idx[bk];
    const float* __restrict__ src = F + ((size_t)b * S_LEN + t) * D_IN;
    float* __restrict__ dst = out + (size_t)bk * D_IN;
    for (int c = threadIdx.x; c < D_IN; c += blockDim.x) dst[c] = src[c];
}

// ---------------------------------------------------------------------------
extern "C" void kernel_launch(void* const* d_in, const int* in_sizes, int n_in,
                              void* d_out, int out_size, void* d_ws, size_t ws_size,
                              hipStream_t stream) {
    const float* F     = (const float*)d_in[0];  // [16,4096,768]
    const float* W1    = (const float*)d_in[1];  // [768,384]
    const float* b1    = (const float*)d_in[2];  // [384]
    const float* gamma = (const float*)d_in[3];  // [384]
    const float* beta  = (const float*)d_in[4];  // [384]
    const float* W2    = (const float*)d_in[5];  // [384]
    const float* b2    = (const float*)d_in[6];  // [1]

    float* tokens = (float*)d_out;                                  // 16*40*768
    float* idx_f  = (float*)d_out + (size_t)B_SZ * KSEL * D_IN;     // 640

    char*    ws     = (char*)d_ws;
    __bf16*  Wp     = (__bf16*)ws;                                  // 589824 B
    float*   scores = (float*)(ws + 589824);                        // 262144 B
    int*     idx_i  = (int*)(ws + 589824 + 262144);                 // 2560 B

    pack_w1<<<(D_IN * H_DIM) / 256, 256, 0, stream>>>(W1, Wp);
    gemm_ln_gelu_score<<<(B_SZ * S_LEN) / 32, 256, 0, stream>>>(
        F, Wp, b1, gamma, beta, W2, b2, scores);
    topk40<<<B_SZ, 256, 0, stream>>>(scores, idx_i, idx_f);
    gather_tokens<<<B_SZ * KSEL, 256, 0, stream>>>(F, idx_i, tokens);
}
